// GroupedQueryAttention_30605936951674
// MI455X (gfx1250) — compile-verified, matrix operand order checked by reading, not run on hardware
//
#include <hip/hip_runtime.h>

// ---------------------------------------------------------------------------
// GQA + RoPE + causal flash attention for MI455X (gfx1250, wave32, WMMA).
// bf16 WMMA (f32 accumulate) everywhere; inputs pre-converted to bf16 once.
// K-tiles staged to LDS via double-buffered global_load_async_to_lds_b128
// (ASYNCcnt) in the attention kernel. Working set fits in 192MB L2.
// ---------------------------------------------------------------------------

typedef __attribute__((ext_vector_type(16))) __bf16 bf16x16;
typedef __attribute__((ext_vector_type(8)))  __bf16 bf16x8;
typedef __attribute__((ext_vector_type(8)))  float  f32x8;

#define BATCH   2
#define S_LEN   2048
#define D_MODEL 1024
#define NHEADS  16
#define NKVH    4
#define HDIM    64
#define MiB     (1024 * 1024)

// ---- WMMA wrapper ----------------------------------------------------------
__device__ __forceinline__ f32x8 wmma_bf16(bf16x16 a, bf16x16 b, f32x8 c) {
  return __builtin_amdgcn_wmma_f32_16x16x32_bf16(false, a, false, b,
                                                 (short)0, c, false, false);
}

// 16-bit A/B fragment (ISA 7.12.2): lane's 16 elems are rowptr[half*8 + 0..7]
// and rowptr[16 + half*8 + 0..7] -> two aligned 16B vector loads (b128).
__device__ __forceinline__ bf16x16 load_frag_bf16(const __bf16* rowptr, int half) {
  const bf16x8 lo = *reinterpret_cast<const bf16x8*>(rowptr + (half << 3));
  const bf16x8 hi = *reinterpret_cast<const bf16x8*>(rowptr + 16 + (half << 3));
  bf16x16 f;
#pragma unroll
  for (int i = 0; i < 8; ++i) { f[i] = lo[i]; f[8 + i] = hi[i]; }
  return f;
}

// Row reductions within a half-wave (C-fragment rows stripe across 16 lanes).
__device__ __forceinline__ float rmax16(float v) {
#pragma unroll
  for (int m = 1; m < 16; m <<= 1) v = fmaxf(v, __shfl_xor(v, m, 16));
  return v;
}
__device__ __forceinline__ float rsum16(float v) {
#pragma unroll
  for (int m = 1; m < 16; m <<= 1) v += __shfl_xor(v, m, 16);
  return v;
}

// ---- CDNA5 async copy: each lane DMAs one 128B row into LDS (ASYNCcnt). ----
// INST_OFFSET applies to both the LDS and global addresses (ISA 10.x async).
__device__ __forceinline__ void async_stage_row128(uint32_t lds_bytes,
                                                   const __bf16* gsrc) {
  asm volatile(
      "global_load_async_to_lds_b128 %0, %1, off\n\t"
      "global_load_async_to_lds_b128 %0, %1, off offset:16\n\t"
      "global_load_async_to_lds_b128 %0, %1, off offset:32\n\t"
      "global_load_async_to_lds_b128 %0, %1, off offset:48\n\t"
      "global_load_async_to_lds_b128 %0, %1, off offset:64\n\t"
      "global_load_async_to_lds_b128 %0, %1, off offset:80\n\t"
      "global_load_async_to_lds_b128 %0, %1, off offset:96\n\t"
      "global_load_async_to_lds_b128 %0, %1, off offset:112"
      :: "v"(lds_bytes), "v"(gsrc) : "memory");
}
__device__ __forceinline__ void wait_async0() {
  asm volatile("s_wait_asynccnt 0x0" ::: "memory");
}

// ---------------------------------------------------------------------------
// Kernel 0: one-time f32 -> bf16 pack (halves L2 bytes, kills per-use cvts).
// ---------------------------------------------------------------------------
__global__ __launch_bounds__(256) void cvt_bf16_kernel(
    const float* __restrict__ in, __bf16* __restrict__ out, int n) {
  const int i = (blockIdx.x * 256 + threadIdx.x) * 8;
  if (i >= n) return;
  f32x8 v = *reinterpret_cast<const f32x8*>(in + i);
  bf16x8 o;
#pragma unroll
  for (int j = 0; j < 8; ++j) o[j] = (__bf16)v[j];
  *reinterpret_cast<bf16x8*>(out + i) = o;
}

// ---------------------------------------------------------------------------
// Kernel 1: q/k/v projections (y = x @ W^T) with fused RoPE on q,k.
// One wave owns one full head's 64 columns (4 N-tiles): 1 A-frag feeds
// 4 WMMAs, and both rotate-half pairs (t,t+2) live in the same lane.
// Grid: x = 32 (8 M-tiles/block), y = 24 head-tasks (16 q, 4 k, 4 v).
// ---------------------------------------------------------------------------
__global__ __launch_bounds__(256) void qkv_rope_kernel(
    const __bf16* __restrict__ xbf, const __bf16* __restrict__ wqbf,
    const __bf16* __restrict__ wkbf, const __bf16* __restrict__ wvbf,
    __bf16* __restrict__ qws, __bf16* __restrict__ kws,
    __bf16* __restrict__ vtws) {
  const int lane = threadIdx.x & 31;
  const int wave = threadIdx.x >> 5;
  const int half = lane >> 4;
  const int l16  = lane & 15;

  const int m0 = (blockIdx.x * 8 + wave) * 16;   // row tile in [B*S, D]
  const int pt = blockIdx.y;

  const __bf16* W;
  int kind, head;
  if (pt < 16)      { kind = 0; head = pt;      W = wqbf; }
  else if (pt < 20) { kind = 1; head = pt - 16; W = wkbf; }
  else              { kind = 2; head = pt - 20; W = wvbf; }

  const __bf16* arow = xbf + (size_t)(m0 + l16) * D_MODEL;
  const __bf16* brow[4];
#pragma unroll
  for (int t = 0; t < 4; ++t)
    brow[t] = W + (size_t)(head * HDIM + t * 16 + l16) * D_MODEL;

  f32x8 acc[4] = {};
#pragma unroll 2
  for (int kk = 0; kk < D_MODEL; kk += 32) {
    __builtin_prefetch(arow + kk + 128, 0, 1);   // global_prefetch_b8
    const bf16x16 a = load_frag_bf16(arow + kk, half);
#pragma unroll
    for (int t = 0; t < 4; ++t) {
      const bf16x16 b = load_frag_bf16(brow[t] + kk, half);
      acc[t] = wmma_bf16(a, b, acc[t]);
    }
  }

  // Epilogue: lane holds col n=l16 of each tile, rows (half*8 + r).
#pragma unroll
  for (int r = 0; r < 8; ++r) {
    const int m = m0 + half * 8 + r;
    const int b = m >> 11;              // S_LEN = 2048
    const int s = m & (S_LEN - 1);
    if (kind == 2) {                    // V: no RoPE, store transposed [HD][S]
      __bf16* vb = vtws + ((size_t)(b * NKVH + head) * HDIM) * S_LEN;
#pragma unroll
      for (int t = 0; t < 4; ++t)
        vb[(size_t)(t * 16 + l16) * S_LEN + s] = (__bf16)acc[t][r];
    } else {
      __bf16* dst = (kind == 0)
          ? qws + ((size_t)(b * NHEADS + head) * S_LEN + s) * HDIM
          : kws + ((size_t)(b * NKVH  + head) * S_LEN + s) * HDIM;
      // RoPE pairs (tile p, tile p+2): d1 = p*16 + l16 in [0,32), same angle
      // for d1 and d1+32.  freq_i = 10000^(-i/32) = exp(-i * ln(1e4)/32).
#pragma unroll
      for (int p = 0; p < 2; ++p) {
        const int d1 = p * 16 + l16;
        const float fr  = __expf(-0.28782313662425576f * (float)d1);
        const float ang = (float)s * fr;
        float sn, cs;
        __sincosf(ang, &sn, &cs);
        const float lo = acc[p][r], hi = acc[p + 2][r];
        dst[d1]      = (__bf16)(lo * cs - hi * sn);
        dst[d1 + 32] = (__bf16)(hi * cs + lo * sn);
      }
    }
  }
}

// ---------------------------------------------------------------------------
// Kernel 2: causal flash attention. One wave owns 16 query rows; keys in
// blocks of 32. K-blocks are double-buffer staged into LDS with
// global_load_async_to_lds_b128 (overlaps DMA with current block's WMMAs).
// ---------------------------------------------------------------------------
__global__ __launch_bounds__(256) void flash_attn_kernel(
    const __bf16* __restrict__ qws, const __bf16* __restrict__ kws,
    const __bf16* __restrict__ vtws, __bf16* __restrict__ ows) {
  __shared__ __align__(16) __bf16 lds_k[8][2][32 * HDIM];  // 2 x 4KB / wave
  __shared__ __align__(16) __bf16 lds_p[8][16 * 32];       // 1KB / wave

  const int lane = threadIdx.x & 31;
  const int wave = threadIdx.x >> 5;
  const int half = lane >> 4;
  const int l16  = lane & 15;

  const int QT = S_LEN / 16;            // 128 q-tiles per (b,h)
  const int gw = blockIdx.x * 8 + wave;
  const int b   = gw / (NHEADS * QT);
  const int rem = gw % (NHEADS * QT);
  const int h   = rem / QT;
  const int q0  = (rem % QT) * 16;
  const int kvh = h >> 2;               // GQA: head h -> kv head h/rep

  const __bf16* qb  = qws  + (size_t)(b * NHEADS + h)   * S_LEN * HDIM;
  const __bf16* kb_ = kws  + (size_t)(b * NKVH   + kvh) * S_LEN * HDIM;
  const __bf16* vtb = vtws + (size_t)(b * NKVH   + kvh) * HDIM * S_LEN;

  // Q A-fragments (contraction over d: two 32-wide K-steps), loaded once.
  const bf16x16 qa0 = load_frag_bf16(qb + (size_t)(q0 + l16) * HDIM + 0,  half);
  const bf16x16 qa1 = load_frag_bf16(qb + (size_t)(q0 + l16) * HDIM + 32, half);

  const float scale = 0.125f;           // 1/sqrt(64)
  float run_max[8], run_sum[8];
  f32x8 o0{}, o1{}, o2{}, o3{};
#pragma unroll
  for (int r = 0; r < 8; ++r) { run_max[r] = -3.0e38f; run_sum[r] = 0.f; }

  const uint32_t kbuf_bytes[2] = {
      (uint32_t)(uintptr_t)&lds_k[wave][0][0] + (uint32_t)lane * 128u,
      (uint32_t)(uintptr_t)&lds_k[wave][1][0] + (uint32_t)lane * 128u};

  const int kend = q0 + 16;
  int buf = 0;
  async_stage_row128(kbuf_bytes[0], kb_ + (size_t)lane * HDIM);  // block 0

  for (int kblk = 0; kblk < kend; kblk += 32) {
    wait_async0();                       // current K block resident in LDS
    if (kblk + 32 < kend)                // prefetch next block into other buf
      async_stage_row128(kbuf_bytes[buf ^ 1],
                         kb_ + (size_t)(kblk + 32 + lane) * HDIM);

    // ---- S = Q K^T for 32 keys (two 16-wide N tiles), K from LDS ----
    const __bf16* kt = &lds_k[wave][buf][0];
    f32x8 s0{}, s1{};
    {
      const __bf16* kr0 = kt + (size_t)(l16) * HDIM;
      const __bf16* kr1 = kt + (size_t)(16 + l16) * HDIM;
      bf16x16 b00 = load_frag_bf16(kr0 + 0,  half);
      bf16x16 b01 = load_frag_bf16(kr0 + 32, half);
      bf16x16 b10 = load_frag_bf16(kr1 + 0,  half);
      bf16x16 b11 = load_frag_bf16(kr1 + 32, half);
      s0 = wmma_bf16(qa0, b00, s0);  s0 = wmma_bf16(qa1, b01, s0);
      s1 = wmma_bf16(qa0, b10, s1);  s1 = wmma_bf16(qa1, b11, s1);
    }

    // ---- scale + causal mask ----
    const bool need_mask = (kblk + 31) > q0;
    const int key0 = kblk + l16, key1 = key0 + 16;
#pragma unroll
    for (int r = 0; r < 8; ++r) {
      if (need_mask) {
        const int qrow = q0 + half * 8 + r;
        s0[r] = (key0 > qrow) ? -1.0e30f : s0[r] * scale;
        s1[r] = (key1 > qrow) ? -1.0e30f : s1[r] * scale;
      } else {
        s0[r] *= scale; s1[r] *= scale;
      }
    }

    // ---- online softmax (rows striped across 16 lanes) ----
#pragma unroll
    for (int r = 0; r < 8; ++r) {
      const float bm = rmax16(fmaxf(s0[r], s1[r]));
      const float nm = fmaxf(run_max[r], bm);
      const float corr = __expf(run_max[r] - nm);
      run_max[r] = nm;
      const float p0 = __expf(s0[r] - nm);
      const float p1 = __expf(s1[r] - nm);
      run_sum[r] = run_sum[r] * corr + rsum16(p0 + p1);
      o0[r] *= corr; o1[r] *= corr; o2[r] *= corr; o3[r] *= corr;
      const int row = half * 8 + r;
      lds_p[wave][row * 32 + l16]      = (__bf16)p0;
      lds_p[wave][row * 32 + 16 + l16] = (__bf16)p1;
    }

    // ---- O += P @ V (contraction over 32 keys, 4 d-tiles) ----
    const bf16x16 pa = load_frag_bf16(&lds_p[wave][l16 * 32], half);
    {
      bf16x16 vb0 = load_frag_bf16(vtb + (size_t)(0  + l16) * S_LEN + kblk, half);
      bf16x16 vb1 = load_frag_bf16(vtb + (size_t)(16 + l16) * S_LEN + kblk, half);
      bf16x16 vb2 = load_frag_bf16(vtb + (size_t)(32 + l16) * S_LEN + kblk, half);
      bf16x16 vb3 = load_frag_bf16(vtb + (size_t)(48 + l16) * S_LEN + kblk, half);
      o0 = wmma_bf16(pa, vb0, o0);
      o1 = wmma_bf16(pa, vb1, o1);
      o2 = wmma_bf16(pa, vb2, o2);
      o3 = wmma_bf16(pa, vb3, o3);
    }
    buf ^= 1;
  }

  // ---- finalize: divide by running sum, store bf16 [B][S][NH*HD] ----
#pragma unroll
  for (int r = 0; r < 8; ++r) {
    const float inv = 1.0f / run_sum[r];
    const int s = q0 + half * 8 + r;
    __bf16* orow = ows + ((size_t)b * S_LEN + s) * D_MODEL + h * HDIM;
    orow[0  + l16] = (__bf16)(o0[r] * inv);
    orow[16 + l16] = (__bf16)(o1[r] * inv);
    orow[32 + l16] = (__bf16)(o2[r] * inv);
    orow[48 + l16] = (__bf16)(o3[r] * inv);
  }
}

// ---------------------------------------------------------------------------
// Kernel 3: output projection out[m][n] = sum_e o[m][e] * Wo[n][e].
// One wave: 16x64 tile (4 accumulators) -> 4 WMMAs per A-fragment.
// ---------------------------------------------------------------------------
__global__ __launch_bounds__(256) void out_proj_kernel(
    const __bf16* __restrict__ ows, const __bf16* __restrict__ wobf,
    float* __restrict__ out) {
  const int lane = threadIdx.x & 31;
  const int wave = threadIdx.x >> 5;
  const int half = lane >> 4;
  const int l16  = lane & 15;

  const int m0 = (blockIdx.x * 8 + wave) * 16;
  const int n0 = blockIdx.y * 64;

  const __bf16* arow = ows + (size_t)(m0 + l16) * D_MODEL;
  const __bf16* brow[4];
#pragma unroll
  for (int t = 0; t < 4; ++t)
    brow[t] = wobf + (size_t)(n0 + t * 16 + l16) * D_MODEL;

  f32x8 acc[4] = {};
#pragma unroll 2
  for (int kk = 0; kk < D_MODEL; kk += 32) {
    __builtin_prefetch(arow + kk + 128, 0, 1);
    const bf16x16 a = load_frag_bf16(arow + kk, half);
#pragma unroll
    for (int t = 0; t < 4; ++t) {
      const bf16x16 bfr = load_frag_bf16(brow[t] + kk, half);
      acc[t] = wmma_bf16(a, bfr, acc[t]);
    }
  }
#pragma unroll
  for (int r = 0; r < 8; ++r) {
    float* orow = out + (size_t)(m0 + half * 8 + r) * D_MODEL + n0;
#pragma unroll
    for (int t = 0; t < 4; ++t) orow[t * 16 + l16] = acc[t][r];
  }
}

// ---------------------------------------------------------------------------
extern "C" void kernel_launch(void* const* d_in, const int* in_sizes, int n_in,
                              void* d_out, int out_size, void* d_ws,
                              size_t ws_size, hipStream_t stream) {
  const float* x  = (const float*)d_in[0];
  const float* Wq = (const float*)d_in[1];
  const float* Wk = (const float*)d_in[2];
  const float* Wv = (const float*)d_in[3];
  const float* Wo = (const float*)d_in[4];
  float* out = (float*)d_out;

  char* ws = (char*)d_ws;
  __bf16* xbf  = (__bf16*)(ws);                            //  8 MiB
  __bf16* wqbf = (__bf16*)(ws + (size_t)8 * MiB);          //  2 MiB
  __bf16* wkbf = (__bf16*)(ws + (size_t)10 * MiB);         // .5 MiB
  __bf16* wvbf = (__bf16*)(ws + (size_t)10 * MiB + 512 * 1024);
  __bf16* wobf = (__bf16*)(ws + (size_t)11 * MiB);         //  2 MiB
  __bf16* qws  = (__bf16*)(ws + (size_t)13 * MiB);         //  8 MiB
  __bf16* kws  = (__bf16*)(ws + (size_t)21 * MiB);         //  2 MiB
  __bf16* vtws = (__bf16*)(ws + (size_t)23 * MiB);         //  2 MiB
  __bf16* ows  = (__bf16*)(ws + (size_t)25 * MiB);         //  8 MiB

  // One-time bf16 packs (grid = n/8/256 blocks).
  cvt_bf16_kernel<<<2048, 256, 0, stream>>>(x,  xbf,  BATCH * S_LEN * D_MODEL);
  cvt_bf16_kernel<<<512,  256, 0, stream>>>(Wq, wqbf, NHEADS * HDIM * D_MODEL);
  cvt_bf16_kernel<<<128,  256, 0, stream>>>(Wk, wkbf, NKVH * HDIM * D_MODEL);
  cvt_bf16_kernel<<<128,  256, 0, stream>>>(Wv, wvbf, NKVH * HDIM * D_MODEL);
  cvt_bf16_kernel<<<512,  256, 0, stream>>>(Wo, wobf, D_MODEL * NHEADS * HDIM);

  qkv_rope_kernel<<<dim3(32, 24), 256, 0, stream>>>(xbf, wqbf, wkbf, wvbf,
                                                    qws, kws, vtws);
  flash_attn_kernel<<<dim3(512), 256, 0, stream>>>(qws, kws, vtws, ows);
  out_proj_kernel<<<dim3(32, 16), 256, 0, stream>>>(ows, wobf, out);
}
